// Text2SemanticDecoder_18056042512703
// MI455X (gfx1250) — matrix operand order repaired and checked by reading, not compiled, not run on hardware
//
#include <hip/hip_runtime.h>
#include <hip/hip_bf16.h>
#include <math.h>

typedef __attribute__((ext_vector_type(16))) _Float16 v16h;
typedef __attribute__((ext_vector_type(8)))  _Float16 v8h;
typedef __attribute__((ext_vector_type(8)))  float    v8f;

#define XLEN   512
#define YLEN   512
#define SEQ    1024
#define DMODEL 512
#define NH     16
#define HDIM   32
#define NLAYER 24
#define FFDIM  2048
#define VOCAB  1025
#define TOPKN  15
#define REPPEN 1.35f

// GEMM macro-tile
#define GM 128
#define GN 64

// ---------------------------------------------------------------------------
// helpers
// ---------------------------------------------------------------------------
__device__ __forceinline__ float half16_max(float v) {
  #pragma unroll
  for (int m = 1; m < 16; m <<= 1) v = fmaxf(v, __shfl_xor(v, m, 32));
  return v;
}
__device__ __forceinline__ float half16_sum(float v) {
  #pragma unroll
  for (int m = 1; m < 16; m <<= 1) v += __shfl_xor(v, m, 32);
  return v;
}
// pack two fp32 into two packed f16 (v_cvt_pk_f16_f32)
__device__ __forceinline__ unsigned pkhalf2(float a, float b) {
  union { _Float16 h[2]; unsigned u; } u_;
  u_.h[0] = (_Float16)a; u_.h[1] = (_Float16)b;
  return u_.u;
}
// 16 contiguous fp32 (as 4 float4) -> v16h halves 0..15
__device__ __forceinline__ v16h cvt16(float4 a, float4 b, float4 c, float4 d) {
  v16h r;
  r[0]  = (_Float16)a.x; r[1]  = (_Float16)a.y; r[2]  = (_Float16)a.z; r[3]  = (_Float16)a.w;
  r[4]  = (_Float16)b.x; r[5]  = (_Float16)b.y; r[6]  = (_Float16)b.z; r[7]  = (_Float16)b.w;
  r[8]  = (_Float16)c.x; r[9]  = (_Float16)c.y; r[10] = (_Float16)c.z; r[11] = (_Float16)c.w;
  r[12] = (_Float16)d.x; r[13] = (_Float16)d.y; r[14] = (_Float16)d.z; r[15] = (_Float16)d.w;
  return r;
}
__device__ __forceinline__ v16h cat8(v8h lo, v8h hi) {
  return __builtin_shufflevector(lo, hi, 0, 1, 2, 3, 4, 5, 6, 7,
                                         8, 9, 10, 11, 12, 13, 14, 15);
}

// ---------------------------------------------------------------------------
// embedding + sinusoidal PE
// ---------------------------------------------------------------------------
__global__ __launch_bounds__(256)
void embed_kernel(const float* __restrict__ x, const int* __restrict__ y,
                  const float* __restrict__ emb, const float* __restrict__ alpha,
                  float* __restrict__ h, float* __restrict__ y_emb_out) {
  int i = blockIdx.x * 256 + threadIdx.x;
  if (i >= SEQ * DMODEL) return;
  int s = i / DMODEL, d = i % DMODEL;
  if (s < XLEN) {
    h[i] = x[i];
  } else {
    int t = s - XLEN;
    int tok = y[t];
    if (tok < 0) tok = 0;
    if (tok >= VOCAB) tok = VOCAB - 1;
    float e = emb[(size_t)tok * DMODEL + d];
    y_emb_out[(size_t)t * DMODEL + d] = e;
    int j = d >> 1;
    float ang = (float)t * __expf(-(float)(2 * j) * (9.210340371976184f / (float)DMODEL));
    float pe = (d & 1) ? __cosf(ang) : __sinf(ang);
    h[i] = e + alpha[0] * pe;
  }
}

// ---------------------------------------------------------------------------
// LDS-tiled WMMA GEMM:  C[M,N] = A[M,K] @ W[N,K]^T + bias  (optional ReLU)
// block = 256 threads (8 waves); macro tile GM x GN = 128 x 64
// wave w computes rows [m0+16w, m0+16w+16) x all 64 cols (4 WMMA subtiles)
// K % 32 == 0, N % 64 == 0, M % 128 == 0
// ---------------------------------------------------------------------------
__global__ __launch_bounds__(256)
void gemm_wmma_kernel(const float* __restrict__ A, const float* __restrict__ W,
                      const float* __restrict__ bias, float* __restrict__ C,
                      int M, int N, int K, int relu) {
  __shared__ alignas(16) _Float16 Ah[GM][40];   // 128x32 f16, padded stride
  __shared__ alignas(16) _Float16 Bh[GN][40];   // 64x32  f16
  const int tid  = threadIdx.x;
  const int wave = tid >> 5;
  const int lane = tid & 31;
  const int hi16 = lane >> 4;
  const int l16  = lane & 15;
  const int m0 = blockIdx.y * GM;
  const int n0 = blockIdx.x * GN;
  (void)M;

  v8f acc[4] = {};

  for (int kk = 0; kk < K; kk += 32) {
    // ---- cooperative global loads (all issued before any conversion) ----
    float4 av[4];
    int rowA[4], c4A[4];
    #pragma unroll
    for (int i = 0; i < 4; ++i) {
      int g = tid + 256 * i;              // 1024 float4 = 128 rows x 8
      rowA[i] = g >> 3; c4A[i] = g & 7;
      av[i] = *(const float4*)(A + (size_t)(m0 + rowA[i]) * K + kk + c4A[i] * 4);
    }
    float4 bw[2];
    int rowB[2], c4B[2];
    #pragma unroll
    for (int i = 0; i < 2; ++i) {
      int g = tid + 256 * i;              // 512 float4 = 64 rows x 8
      rowB[i] = g >> 3; c4B[i] = g & 7;
      bw[i] = *(const float4*)(W + (size_t)(n0 + rowB[i]) * K + kk + c4B[i] * 4);
    }
    // unconditional speculative prefetch of next K-slab (branch-free;
    // dropped silently on translation failure per ISA 10.5)
    __builtin_prefetch(A + (size_t)(m0 + rowA[0]) * K + kk + c4A[0] * 4 + 32, 0, 1);
    __builtin_prefetch(W + (size_t)(n0 + rowB[0]) * K + kk + c4B[0] * 4 + 32, 0, 1);

    // ---- convert + stage to LDS ----
    #pragma unroll
    for (int i = 0; i < 4; ++i) {
      unsigned* p = (unsigned*)&Ah[rowA[i]][c4A[i] * 4];
      p[0] = pkhalf2(av[i].x, av[i].y);
      p[1] = pkhalf2(av[i].z, av[i].w);
    }
    #pragma unroll
    for (int i = 0; i < 2; ++i) {
      unsigned* p = (unsigned*)&Bh[rowB[i]][c4B[i] * 4];
      p[0] = pkhalf2(bw[i].x, bw[i].y);
      p[1] = pkhalf2(bw[i].z, bw[i].w);
    }
    __syncthreads();
    // ---- fragments from LDS (b128 DS loads) + 4 WMMAs ----
    const _Float16* ar = &Ah[wave * 16 + l16][0];
    v16h a = cat8(*(const v8h*)(ar + 8 * hi16),
                  *(const v8h*)(ar + 8 * hi16 + 16));
    #pragma unroll
    for (int j = 0; j < 4; ++j) {
      const _Float16* br = &Bh[j * 16 + l16][0];
      v16h b = cat8(*(const v8h*)(br + 8 * hi16),
                    *(const v8h*)(br + 8 * hi16 + 16));
      acc[j] = __builtin_amdgcn_wmma_f32_16x16x32_f16(
          false, a, false, b, (short)0, acc[j], false, false);
    }
    __syncthreads();
  }
  // ---- epilogue ----
  #pragma unroll
  for (int j = 0; j < 4; ++j) {
    int col = n0 + j * 16 + l16;
    float bv = bias ? bias[col] : 0.f;
    #pragma unroll
    for (int r = 0; r < 8; ++r) {
      int row = m0 + wave * 16 + r + hi16 * 8;
      float v = acc[j][r] + bv;
      if (relu) v = fmaxf(v, 0.f);
      C[(size_t)row * N + col] = v;
    }
  }
}

// ---------------------------------------------------------------------------
// copy K/V slices of qkv buffer into the k/v caches (in d_out)
// ---------------------------------------------------------------------------
__global__ __launch_bounds__(256)
void copy_kv_kernel(const float* __restrict__ qkv, float* __restrict__ kout,
                    float* __restrict__ vout) {
  int i = blockIdx.x * 256 + threadIdx.x;
  if (i >= SEQ * DMODEL) return;
  int s = i / DMODEL, d = i % DMODEL;
  const float* row = qkv + (size_t)s * (3 * DMODEL);
  kout[i] = row[DMODEL + d];
  vout[i] = row[2 * DMODEL + d];
}

// ---------------------------------------------------------------------------
// flash attention: one wave per (head, 16-row q tile); block-causal mask
// ---------------------------------------------------------------------------
__global__ __launch_bounds__(32)
void attn_kernel(const float* __restrict__ qkv, float* __restrict__ o_out) {
  __shared__ alignas(16) _Float16 pS[16][40];
  const int lane = threadIdx.x & 31;
  const int hi16 = lane >> 4;
  const int l16  = lane & 15;
  const int head = blockIdx.x;
  const int qbase = blockIdx.y * 16;
  const bool isY = (qbase >= XLEN);
  const int kmax = isY ? (qbase + 16) : XLEN;   // keys in [0, kmax), kmax <= SEQ
  const float scale = 0.17677669529663687f;     // 1/sqrt(32)

  // Q A-fragment (fixed for whole loop)
  v16h aq;
  {
    const float* qp = qkv + (size_t)(qbase + l16) * (3 * DMODEL) + head * HDIM + 16 * hi16;
    aq = cvt16(*(const float4*)(qp),      *(const float4*)(qp + 4),
               *(const float4*)(qp + 8),  *(const float4*)(qp + 12));
  }

  float rmax[8], rsum[8];
  #pragma unroll
  for (int r = 0; r < 8; ++r) { rmax[r] = -1e30f; rsum[r] = 0.f; }
  v8f o0 = {}, o1 = {};

  for (int kb = 0; kb < kmax; kb += 32) {
    // ---- load both K sub-tiles (8 independent float4 loads), then convert ----
    const float* kp0 = qkv + (size_t)(kb + l16) * (3 * DMODEL) + DMODEL + head * HDIM + 16 * hi16;
    const float* kp1 = kp0 + (size_t)16 * (3 * DMODEL);
    float4 kf[8];
    #pragma unroll
    for (int i = 0; i < 4; ++i) kf[i]     = *(const float4*)(kp0 + 4 * i);
    #pragma unroll
    for (int i = 0; i < 4; ++i) kf[4 + i] = *(const float4*)(kp1 + 4 * i);
    v16h bk0 = cvt16(kf[0], kf[1], kf[2], kf[3]);
    v16h bk1 = cvt16(kf[4], kf[5], kf[6], kf[7]);
    v8f z = {};
    v8f s0 = __builtin_amdgcn_wmma_f32_16x16x32_f16(false, aq, false, bk0, (short)0, z, false, false);
    v8f s1 = __builtin_amdgcn_wmma_f32_16x16x32_f16(false, aq, false, bk1, (short)0, z, false, false);

    // ---- V tile: batch all 32 strided loads, then convert ----
    float vf[2][16];
    #pragma unroll
    for (int j = 0; j < 2; ++j)
      #pragma unroll
      for (int hi = 0; hi < 16; ++hi)
        vf[j][hi] = qkv[(size_t)(kb + 16 * hi16 + hi) * (3 * DMODEL) + 2 * DMODEL +
                        head * HDIM + 16 * j + l16];

    // ---- scale + mask + online softmax update ----
    #pragma unroll
    for (int r = 0; r < 8; ++r) {
      float v0 = s0[r] * scale;
      float v1 = s1[r] * scale;
      if (isY) {
        int qi = qbase + r + hi16 * 8;
        if (kb + l16 > qi)      v0 = -1e30f;
        if (kb + 16 + l16 > qi) v1 = -1e30f;
      }
      float cmax = half16_max(fmaxf(v0, v1));
      float nm   = fmaxf(rmax[r], cmax);
      float corr = __expf(rmax[r] - nm);
      rmax[r] = nm;
      float p0 = __expf(v0 - nm);
      float p1 = __expf(v1 - nm);
      s0[r] = p0; s1[r] = p1;
      rsum[r] = rsum[r] * corr + half16_sum(p0 + p1);
      o0[r] *= corr;
      o1[r] *= corr;
    }
    // ---- stage P (C layout) -> LDS -> A layout (b128 DS loads) ----
    #pragma unroll
    for (int r = 0; r < 8; ++r) {
      pS[r + 8 * hi16][l16]      = (_Float16)s0[r];
      pS[r + 8 * hi16][16 + l16] = (_Float16)s1[r];
    }
    __syncthreads();
    const _Float16* pr = &pS[l16][0];
    v16h ap = cat8(*(const v8h*)(pr + 16 * hi16),
                   *(const v8h*)(pr + 16 * hi16 + 8));
    // ---- O += P x V ----
    v16h bv0, bv1;
    #pragma unroll
    for (int hi = 0; hi < 16; ++hi) { bv0[hi] = (_Float16)vf[0][hi]; bv1[hi] = (_Float16)vf[1][hi]; }
    o0 = __builtin_amdgcn_wmma_f32_16x16x32_f16(false, ap, false, bv0, (short)0, o0, false, false);
    o1 = __builtin_amdgcn_wmma_f32_16x16x32_f16(false, ap, false, bv1, (short)0, o1, false, false);
    __syncthreads();
  }
  // ---- epilogue ----
  #pragma unroll
  for (int r = 0; r < 8; ++r) {
    int row = qbase + r + 8 * hi16;
    float inv = 1.f / rsum[r];
    o_out[(size_t)row * DMODEL + head * HDIM + l16]      = o0[r] * inv;
    o_out[(size_t)row * DMODEL + head * HDIM + 16 + l16] = o1[r] * inv;
  }
}

// ---------------------------------------------------------------------------
// h = LayerNorm(h + delta) * w + b   (one block per row, D = 512, block = 256)
// ---------------------------------------------------------------------------
__global__ __launch_bounds__(256)
void add_ln_kernel(float* __restrict__ h, const float* __restrict__ delta,
                   const float* __restrict__ w, const float* __restrict__ b) {
  __shared__ float red[256];
  __shared__ float s_mu, s_rstd;
  int row = blockIdx.x, t = threadIdx.x;
  size_t i0 = (size_t)row * DMODEL + t, i1 = i0 + 256;
  float x0 = h[i0] + delta[i0];
  float x1 = h[i1] + delta[i1];
  red[t] = x0 + x1;
  __syncthreads();
  for (int s = 128; s > 0; s >>= 1) { if (t < s) red[t] += red[t + s]; __syncthreads(); }
  if (t == 0) s_mu = red[0] * (1.f / DMODEL);
  __syncthreads();
  float mu = s_mu, d0 = x0 - mu, d1 = x1 - mu;
  red[t] = d0 * d0 + d1 * d1;
  __syncthreads();
  for (int s = 128; s > 0; s >>= 1) { if (t < s) red[t] += red[t + s]; __syncthreads(); }
  if (t == 0) s_rstd = rsqrtf(red[0] * (1.f / DMODEL) + 1e-5f);
  __syncthreads();
  float rs = s_rstd;
  h[i0] = d0 * rs * w[t] + b[t];
  h[i1] = d1 * rs * w[t + 256] + b[t + 256];
}

// ---------------------------------------------------------------------------
// logits[v] = h[S-1] . pred_w[v] + pred_b[v]
// ---------------------------------------------------------------------------
__global__ __launch_bounds__(128)
void logits_kernel(const float* __restrict__ h, const float* __restrict__ pw,
                   const float* __restrict__ pb, float* __restrict__ lg) {
  __shared__ float red[128];
  int v = blockIdx.x, t = threadIdx.x;
  const float* hr = h + (size_t)(SEQ - 1) * DMODEL;
  const float* wr = pw + (size_t)v * DMODEL;
  float acc = 0.f;
  for (int i = t; i < DMODEL; i += 128) acc += hr[i] * wr[i];
  red[t] = acc;
  __syncthreads();
  for (int s = 64; s > 0; s >>= 1) { if (t < s) red[t] += red[t + s]; __syncthreads(); }
  if (t == 0) lg[v] = red[0] + pb[v];
}

// ---------------------------------------------------------------------------
// repetition penalty + top-15 filter + softmax + sample
// ---------------------------------------------------------------------------
__global__ __launch_bounds__(1024)
void sample_kernel(const float* __restrict__ lg_in, const int* __restrict__ y,
                   float* __restrict__ probs_out, float* __restrict__ ynew_out) {
  __shared__ float slg[VOCAB];
  __shared__ float swork[VOCAB];
  __shared__ int   sflag[VOCAB];
  __shared__ float rv[1024];
  __shared__ int   ri[1024];
  __shared__ float s_pivot, s_max, s_sum;
  __shared__ int   s_amax;
  int t = threadIdx.x;

  for (int i = t; i < VOCAB; i += 1024) { slg[i] = lg_in[i]; sflag[i] = 0; }
  __syncthreads();
  for (int i = t; i < YLEN; i += 1024) {
    int tok = y[i];
    if (tok >= 0 && tok < VOCAB) sflag[tok] = 1;
  }
  __syncthreads();
  for (int i = t; i < VOCAB; i += 1024) {
    float v = slg[i];
    if (sflag[i]) v = (v < 0.f) ? v * REPPEN : v / REPPEN;
    slg[i] = v;
    swork[i] = v;
  }
  __syncthreads();

  for (int it = 0; it < TOPKN; ++it) {
    float bv = -1e30f; int bi = 0x7fffffff;
    for (int i = t; i < VOCAB; i += 1024)
      if (swork[i] > bv || (swork[i] == bv && i < bi)) { bv = swork[i]; bi = i; }
    rv[t] = bv; ri[t] = bi;
    __syncthreads();
    for (int s = 512; s > 0; s >>= 1) {
      if (t < s) {
        if (rv[t + s] > rv[t] || (rv[t + s] == rv[t] && ri[t + s] < ri[t])) {
          rv[t] = rv[t + s]; ri[t] = ri[t + s];
        }
      }
      __syncthreads();
    }
    if (t == 0) {
      if (it == 0) { s_max = rv[0]; s_amax = ri[0]; }
      s_pivot = rv[0];
      swork[ri[0]] = -1e30f;
    }
    __syncthreads();
  }

  float pivot = s_pivot, mx = s_max;
  float part = 0.f;
  for (int i = t; i < VOCAB; i += 1024) {
    float v = slg[i];
    float e = (v >= pivot) ? __expf(v - mx) : 0.f;
    swork[i] = e;
    part += e;
  }
  rv[t] = part;
  __syncthreads();
  for (int s = 512; s > 0; s >>= 1) { if (t < s) rv[t] += rv[t + s]; __syncthreads(); }
  if (t == 0) s_sum = rv[0];
  __syncthreads();
  float inv = 1.f / s_sum;
  for (int i = t; i < VOCAB; i += 1024) probs_out[i] = swork[i] * inv;
  for (int i = t; i < YLEN; i += 1024) ynew_out[i] = (float)y[i];
  if (t == 0) ynew_out[YLEN] = (float)s_amax;
}

// ---------------------------------------------------------------------------
// host-side orchestration
// ---------------------------------------------------------------------------
extern "C" void kernel_launch(void* const* d_in, const int* in_sizes, int n_in,
                              void* d_out, int out_size, void* d_ws, size_t ws_size,
                              hipStream_t stream) {
  (void)in_sizes; (void)n_in; (void)out_size; (void)ws_size;
  const float* x       = (const float*)d_in[0];
  const int*   y       = (const int*)d_in[1];
  const float* emb     = (const float*)d_in[2];
  const float* alpha   = (const float*)d_in[3];
  const float* qkv_w   = (const float*)d_in[4];
  const float* qkv_b   = (const float*)d_in[5];
  const float* out_w   = (const float*)d_in[6];
  const float* out_b   = (const float*)d_in[7];
  const float* w1      = (const float*)d_in[8];
  const float* b1      = (const float*)d_in[9];
  const float* w2      = (const float*)d_in[10];
  const float* b2      = (const float*)d_in[11];
  const float* ln1_w   = (const float*)d_in[12];
  const float* ln1_b   = (const float*)d_in[13];
  const float* ln2_w   = (const float*)d_in[14];
  const float* ln2_b   = (const float*)d_in[15];
  const float* pred_w  = (const float*)d_in[16];
  const float* pred_b  = (const float*)d_in[17];

  float* out = (float*)d_out;
  float* ws  = (float*)d_ws;

  // workspace (floats): h | bufA | bufB | logits  (~12.6 MB)
  float* h    = ws;
  float* bufA = h    + (size_t)SEQ * DMODEL;
  float* bufB = bufA + (size_t)SEQ * FFDIM;
  float* lg   = bufB + (size_t)SEQ * DMODEL;

  // output (floats): y_new(513) | probs(1025) | k_cache | v_cache | y_emb
  const size_t KV = (size_t)NLAYER * SEQ * DMODEL;
  float* o_ynew  = out;
  float* o_probs = out + 513;
  float* o_k     = out + 1538;
  float* o_v     = o_k + KV;
  float* o_yemb  = o_v + KV;

  embed_kernel<<<(SEQ * DMODEL) / 256, 256, 0, stream>>>(x, y, emb, alpha, h, o_yemb);

  for (int l = 0; l < NLAYER; ++l) {
    gemm_wmma_kernel<<<dim3((3 * DMODEL) / GN, SEQ / GM), 256, 0, stream>>>(
        h, qkv_w + (size_t)l * 3 * DMODEL * DMODEL, qkv_b + (size_t)l * 3 * DMODEL,
        bufA, SEQ, 3 * DMODEL, DMODEL, 0);
    copy_kv_kernel<<<(SEQ * DMODEL) / 256, 256, 0, stream>>>(
        bufA, o_k + (size_t)l * SEQ * DMODEL, o_v + (size_t)l * SEQ * DMODEL);
    attn_kernel<<<dim3(NH, SEQ / 16), 32, 0, stream>>>(bufA, bufB);
    gemm_wmma_kernel<<<dim3(DMODEL / GN, SEQ / GM), 256, 0, stream>>>(
        bufB, out_w + (size_t)l * DMODEL * DMODEL, out_b + (size_t)l * DMODEL,
        bufA, SEQ, DMODEL, DMODEL, 0);
    add_ln_kernel<<<SEQ, 256, 0, stream>>>(h, bufA,
        ln1_w + (size_t)l * DMODEL, ln1_b + (size_t)l * DMODEL);
    gemm_wmma_kernel<<<dim3(FFDIM / GN, SEQ / GM), 256, 0, stream>>>(
        h, w1 + (size_t)l * FFDIM * DMODEL, b1 + (size_t)l * FFDIM,
        bufA, SEQ, FFDIM, DMODEL, 1);
    gemm_wmma_kernel<<<dim3(DMODEL / GN, SEQ / GM), 256, 0, stream>>>(
        bufA, w2 + (size_t)l * DMODEL * FFDIM, b2 + (size_t)l * DMODEL,
        bufB, SEQ, DMODEL, FFDIM, 0);
    add_ln_kernel<<<SEQ, 256, 0, stream>>>(h, bufB,
        ln2_w + (size_t)l * DMODEL, ln2_b + (size_t)l * DMODEL);
  }

  logits_kernel<<<VOCAB, 128, 0, stream>>>(h, pred_w, pred_b, lg);
  sample_kernel<<<1, 1024, 0, stream>>>(lg, y, o_probs, o_ynew);
}
